// MultiHeadAttention_2791728742796
// MI455X (gfx1250) — matrix operand order repaired and checked by reading, not compiled, not run on hardware
//
#include <hip/hip_runtime.h>
#include <hip/hip_bf16.h>

// ---- sizes (compile-time constants from the reference) ----
#define BB   32
#define TT   4096
#define DD   1024      // INPUT_DIM == ATTN_DIM == VALUES_DIM
#define HH   16
#define DK   64        // per-head dim
#define TW   8         // workgroups per batch in the fused pass
#define CT   512       // t-rows per workgroup (TT / TW)
#define CH   128       // t-rows per chunk (8 waves x 16)

typedef __attribute__((ext_vector_type(16))) __bf16 v16bf;
typedef __attribute__((ext_vector_type(8)))  float  v8f;

// Native fptrunc -> backend lowers to v_cvt_pk_bf16_f32 (1 op per 2 elements)
__device__ __forceinline__ __bf16 f2bf(float f) { return (__bf16)f; }
__device__ __forceinline__ unsigned short f2bfu(float f) {
    __bf16 h = (__bf16)f;
    return __builtin_bit_cast(unsigned short, h);
}
__device__ __forceinline__ __bf16 u2bf(unsigned short s) {
    return __builtin_bit_cast(__bf16, s);
}

// ---------------------------------------------------------------------------
// Stage 0a: qh[b, h*64+d] = (vector[b,:] @ Wq[:, a] + bq[a]) / sqrt(64)
// ---------------------------------------------------------------------------
__global__ void k_qh(const float* __restrict__ vec, const float* __restrict__ Wq,
                     const float* __restrict__ bq, float* __restrict__ qh) {
    int idx = blockIdx.x * blockDim.x + threadIdx.x;   // [0, 32768)
    int b = idx >> 10, a = idx & 1023;
    const float* v = vec + (size_t)b * DD;
    float s = bq[a];
    for (int c = 0; c < DD; ++c) s += v[c] * Wq[(size_t)c * DD + a];
    qh[idx] = s * 0.125f;                              // scale = sqrt(1024/16) = 8
}

// ---------------------------------------------------------------------------
// Stage 0b: wkq_bf16[b,h,c] = bf16( sum_d Wk[c, h*64+d] * qh[b, h*64+d] )
// Collapses Q @ K^T through the K projection; stored pre-converted to bf16
// so the fused pass never converts its B panel.
// ---------------------------------------------------------------------------
__global__ void k_wkq(const float* __restrict__ Wk, const float* __restrict__ qh,
                      unsigned short* __restrict__ wkqb) {
    int idx = blockIdx.x * blockDim.x + threadIdx.x;   // b*16384 + h*1024 + c
    int b = idx >> 14;
    int hc = idx & 16383;
    int h = hc >> 10, c = hc & 1023;
    const float* q = qh + (size_t)b * DD + h * DK;
    const float* w = Wk + (size_t)c * DD + h * DK;
    float s = 0.f;
#pragma unroll 8
    for (int d = 0; d < DK; ++d) s += w[d] * q[d];
    wkqb[idx] = f2bfu(s);
}

// Stage 0c: sbias[b,h] = qh[b,h,:] . bk[h*64:...]  (bk folded into scores)
__global__ void k_sbias(const float* __restrict__ qh, const float* __restrict__ bk,
                        float* __restrict__ sbias) {
    int idx = blockIdx.x * blockDim.x + threadIdx.x;   // [0, 512)
    int b = idx >> 4, h = idx & 15;
    const float* q = qh + (size_t)b * DD + h * DK;
    const float* kb = bk + h * DK;
    float s = 0.f;
#pragma unroll 8
    for (int d = 0; d < DK; ++d) s += q[d] * kb[d];
    sbias[idx] = s;
}

// ---------------------------------------------------------------------------
// Fused flash pass: one HBM sweep over `matrix`.
// Grid = 32 b * 8 t-slices = 256 blocks, 8 waves each.
// Per 128-row chunk:
//   (1) each wave: 16t x 16h scores tile via WMMA (B panel = wkq bf16 in LDS,
//       staged once per block with global_load_async_to_lds_b128)
//       -> raw scores to global, masked scores to LDS
//   (2) running max/sum per head; p = exp(s - M) written to LDS as bf16 (P^T)
//   (3) each wave: rescale its f32 ctx accumulators by exp(M_old - M_new) and
//       accumulate ctx += P^T @ matrix via WMMA (matrix chunk re-read, L2-hot)
// Emits per-WG partials (ctx, M, S) for the cross-slice reduction.
// ---------------------------------------------------------------------------
__global__ void __launch_bounds__(256)
k_flash(const float* __restrict__ matrix, const int* __restrict__ mask,
        const unsigned short* __restrict__ wkqb, const float* __restrict__ sbias,
        float* __restrict__ scores, float* __restrict__ ctxp,
        float* __restrict__ mpart, float* __restrict__ spart) {
    int b    = blockIdx.x >> 3;
    int tw   = blockIdx.x & 7;
    int tid  = threadIdx.x;
    int wave = tid >> 5, lane = tid & 31;
    int ln   = lane & 15, khalf = lane >> 4;

    __shared__ __align__(16) unsigned short sWkq[HH * DD];  // 32 KB, [h][k] bf16
    __shared__ float          sS[CH][HH];                   // 8 KB masked scores [t][h]
    __shared__ __align__(16) unsigned short sP[HH][CH];     // 4 KB bf16 P^T [h][t]
    __shared__ float sMax[HH], sSum[HH], sR[HH];

    // ---- async-stage the 32 KB wkq bf16 panel into LDS (CDNA5 async path) ----
    {
        const unsigned short* src = wkqb + (size_t)b * HH * DD;
        unsigned lds0 = (unsigned)(size_t)(&sWkq[0]);
        unsigned long long g0 = (unsigned long long)(size_t)src;
#pragma unroll
        for (int i = 0; i < 8; ++i) {
            unsigned off = (unsigned)(tid + i * 256) * 16u;  // 16B per lane-op
            unsigned ldsa = lds0 + off;
            unsigned long long ga = g0 + off;
            asm volatile("global_load_async_to_lds_b128 %0, %1, off"
                         :: "v"(ldsa), "v"(ga) : "memory");
        }
        asm volatile("s_wait_asynccnt 0x0" ::: "memory");
    }
    if (tid < HH) { sMax[tid] = -3.0e38f; sSum[tid] = 0.f; }
    __syncthreads();

    float sb = sbias[b * HH + ln];
    v8f acc[8] = {};                      // ctx accumulators: 8 c-tiles x (16h x 16c)
    int c0 = wave * 128;                  // this wave's 128-column c-range
    const int tbase = tw * CT;

    for (int ch = 0; ch < CT / CH; ++ch) {
        int tch = tbase + ch * CH;
        // ---- (1) scores tile: rows trow..trow+15, all 16 heads ----
        {
            int trow = tch + wave * 16;
            const float* arow = matrix + ((size_t)(b * TT + trow + ln)) * DD;
            v8f sc = {};
            for (int k0 = 0; k0 < DD; k0 += 32) {
                const float* pa = arow + k0 + khalf * 16;
                __builtin_prefetch(pa + 64, 0, 0);
                v16bf A;
#pragma unroll
                for (int j = 0; j < 8; ++j) { A[j] = f2bf(pa[j]); A[8 + j] = f2bf(pa[8 + j]); }
                const v16bf Bm =
                    *reinterpret_cast<const v16bf*>(&sWkq[ln * DD + k0 + khalf * 16]);
                sc = __builtin_amdgcn_wmma_f32_16x16x32_bf16(false, A, false, Bm,
                                                             (short)0, sc, false, false);
            }
            // D: lane -> t = trow + khalf*8 + r ; head = ln
            int tloc0 = wave * 16 + khalf * 8;
            float* o = scores + ((size_t)(b * HH + ln)) * TT + trow + khalf * 8;
            const int* mk = mask + (size_t)b * TT + trow + khalf * 8;
#pragma unroll
            for (int r = 0; r < 8; ++r) {
                float x = sc[r] + sb;
                o[r] = x;                                       // raw scores out
                sS[tloc0 + r][ln] = (mk[r] > 0) ? x : -1.0e30f; // masked for softmax
            }
        }
        __syncthreads();
        // ---- (2) online softmax stats + bf16 P^T ----
        {
            int h = tid >> 4;            // 16 threads per head
            int sl = tid & 15;
            float vv[8];
            float lm = -3.0e38f;
#pragma unroll
            for (int j = 0; j < 8; ++j) { vv[j] = sS[sl + j * 16][h]; lm = fmaxf(lm, vv[j]); }
#pragma unroll
            for (int o2 = 8; o2 > 0; o2 >>= 1) lm = fmaxf(lm, __shfl_xor(lm, o2, 16));
            float mold = sMax[h];
            float mnew = fmaxf(mold, lm);
            float ls = 0.f;
#pragma unroll
            for (int j = 0; j < 8; ++j) {
                float p = __expf(vv[j] - mnew);
                ls += p;
                sP[h][sl + j * 16] = f2bfu(p);
            }
#pragma unroll
            for (int o2 = 8; o2 > 0; o2 >>= 1) ls += __shfl_xor(ls, o2, 16);
            if (sl == 0) {
                sR[h]   = __expf(mold - mnew);
                sSum[h] = sSum[h] * sR[h] + ls;
                sMax[h] = mnew;
            }
        }
        __syncthreads();
        // ---- (3) rescale accumulators, ctx += P^T @ matrix chunk ----
        {
            float f[8];
#pragma unroll
            for (int r = 0; r < 8; ++r) f[r] = sR[khalf * 8 + r];
#pragma unroll
            for (int ct = 0; ct < 8; ++ct)
#pragma unroll
                for (int r = 0; r < 8; ++r) acc[ct][r] *= f[r];

            for (int ks = 0; ks < CH / 32; ++ks) {
                // A: P^T bf16 from LDS, row h = ln, K(t) = ks*32 + layout offsets
                const unsigned short* pa = &sP[ln][ks * 32 + khalf * 16];
                v16bf A;
#pragma unroll
                for (int j = 0; j < 8; ++j) {
                    A[j]     = u2bf(pa[j]);
                    A[8 + j] = u2bf(pa[8 + j]);
                }
                const float* pbb =
                    matrix + ((size_t)(b * TT + tch + ks * 32 + khalf * 16)) * DD + c0 + ln;
#pragma unroll
                for (int ct = 0; ct < 8; ++ct) {
                    const float* pb = pbb + ct * 16;
                    v16bf Bm;
#pragma unroll
                    for (int j = 0; j < 16; ++j) Bm[j] = f2bf(pb[(size_t)j * DD]);
                    acc[ct] = __builtin_amdgcn_wmma_f32_16x16x32_bf16(false, A, false, Bm,
                                                                      (short)0, acc[ct],
                                                                      false, false);
                }
            }
        }
        __syncthreads();
    }
    // ---- write per-WG partials ----
    int wg = b * TW + tw;
#pragma unroll
    for (int ct = 0; ct < 8; ++ct) {
        int c = c0 + ct * 16 + ln;
#pragma unroll
        for (int r = 0; r < 8; ++r)
            ctxp[((size_t)wg * HH + khalf * 8 + r) * DD + c] = acc[ct][r];
    }
    if (tid < HH) { mpart[wg * HH + tid] = sMax[tid]; spart[wg * HH + tid] = sSum[tid]; }
}

// ---------------------------------------------------------------------------
// Cross-slice softmax stats: global M, 1/S, and per-slice combine scales.
// ---------------------------------------------------------------------------
__global__ void k_stats(const float* __restrict__ mpart, const float* __restrict__ spart,
                        float* __restrict__ escale, float* __restrict__ gM,
                        float* __restrict__ gInvS) {
    int bh = blockIdx.x * blockDim.x + threadIdx.x;    // [0, 512)
    int b = bh >> 4, h = bh & 15;
    float M = -3.0e38f;
#pragma unroll
    for (int w = 0; w < TW; ++w) M = fmaxf(M, mpart[(b * TW + w) * HH + h]);
    float e[TW];
    float S = 0.f;
#pragma unroll
    for (int w = 0; w < TW; ++w) {
        e[w] = __expf(mpart[(b * TW + w) * HH + h] - M);
        S += spart[(b * TW + w) * HH + h] * e[w];
    }
    float inv = 1.0f / S;
#pragma unroll
    for (int w = 0; w < TW; ++w) escale[(b * TW + w) * HH + h] = e[w] * inv;
    gM[bh] = M;
    gInvS[bh] = inv;
}

// attn[b,h,t] = exp(masked_score - M) / S   (scores re-read, L2-resident)
__global__ void k_attn(const float* __restrict__ scores, const int* __restrict__ mask,
                       const float* __restrict__ gM, const float* __restrict__ gInvS,
                       float* __restrict__ attn) {
    int idx = blockIdx.x * blockDim.x + threadIdx.x;   // [0, 2097152)
    int bh = idx >> 12;
    int t  = idx & 4095;
    int b  = bh >> 4;
    float x = (mask[(size_t)b * TT + t] > 0) ? scores[idx] : -1.0e30f;
    attn[idx] = __expf(x - gM[bh]) * gInvS[bh];
}

// ctx[b,h,c] = sum_w ctxp_w * escale_w
__global__ void k_combine(const float* __restrict__ ctxp, const float* __restrict__ escale,
                          float* __restrict__ ctx) {
    int idx = blockIdx.x * blockDim.x + threadIdx.x;   // b*16384 + h*1024 + c
    int b = idx >> 14;
    int h = (idx >> 10) & 15;
    int c = idx & 1023;
    float s = 0.f;
#pragma unroll
    for (int w = 0; w < TW; ++w)
        s += ctxp[((size_t)(b * TW + w) * HH + h) * DD + c] * escale[(b * TW + w) * HH + h];
    ctx[idx] = s;
}

// ---------------------------------------------------------------------------
// Stage 3: pre[b, h*64+d] = ctx[b,h,:] @ Wv[:, h*64+d] + bv   (sum attn == 1)
// ---------------------------------------------------------------------------
__global__ void k_vo(const float* __restrict__ ctx, const float* __restrict__ Wv,
                     const float* __restrict__ bv, float* __restrict__ pre) {
    int idx = blockIdx.x * blockDim.x + threadIdx.x;   // [0, 32768)
    int b = idx >> 10, a = idx & 1023;
    int h = a >> 6;
    const float* cx = ctx + ((size_t)(b * HH + h)) * DD;
    float s = bv[a];
    for (int c = 0; c < DD; ++c) s += cx[c] * Wv[(size_t)c * DD + a];
    pre[idx] = s;
}

// Stage 4: out[b,o] = pre[b,:] @ Wo[:,o] + bo[o]
__global__ void k_out(const float* __restrict__ pre, const float* __restrict__ Wo,
                      const float* __restrict__ bo, float* __restrict__ out) {
    int idx = blockIdx.x * blockDim.x + threadIdx.x;   // [0, 32768)
    int b = idx >> 10, o = idx & 1023;
    const float* p = pre + (size_t)b * DD;
    float s = bo[o];
    for (int a = 0; a < DD; ++a) s += p[a] * Wo[(size_t)a * DD + o];
    out[idx] = s;
}

// ---------------------------------------------------------------------------
extern "C" void kernel_launch(void* const* d_in, const int* in_sizes, int n_in,
                              void* d_out, int out_size, void* d_ws, size_t ws_size,
                              hipStream_t stream) {
    (void)in_sizes; (void)n_in; (void)out_size; (void)ws_size;
    const float* vector = (const float*)d_in[0];
    const float* matrix = (const float*)d_in[1];
    const int*   mask   = (const int*)  d_in[2];
    const float* Wq     = (const float*)d_in[3];
    const float* bq     = (const float*)d_in[4];
    const float* Wk     = (const float*)d_in[5];
    const float* bk     = (const float*)d_in[6];
    const float* Wv     = (const float*)d_in[7];
    const float* bv     = (const float*)d_in[8];
    const float* Wo     = (const float*)d_in[9];
    const float* bo     = (const float*)d_in[10];

    float* out    = (float*)d_out;                       // [32, 1024]
    float* attn   = out  + (size_t)BB * DD;              // [32, 16, 4096]
    float* scores = attn + (size_t)BB * HH * TT;         // [32, 16, 4096]

    // workspace layout (~21 MB)
    float* ws     = (float*)d_ws;
    float* qh     = ws;                                       // 32768
    float* sbias  = qh     + (size_t)BB * DD;                 // 512
    float* mpart  = sbias  + (size_t)BB * HH;                 // 4096
    float* spart  = mpart  + (size_t)BB * TW * HH;            // 4096
    float* escale = spart  + (size_t)BB * TW * HH;            // 4096
    float* gM     = escale + (size_t)BB * TW * HH;            // 512
    float* gInvS  = gM     + (size_t)BB * HH;                 // 512
    float* ctx    = gInvS  + (size_t)BB * HH;                 // 524288
    float* pre    = ctx    + (size_t)BB * HH * DD;            // 32768
    float* ctxp   = pre    + (size_t)BB * DD;                 // 4194304
    unsigned short* wkqb = (unsigned short*)(ctxp + (size_t)BB * TW * HH * DD); // 524288 u16

    k_qh     <<<128,  256, 0, stream>>>(vector, Wq, bq, qh);
    k_wkq    <<<2048, 256, 0, stream>>>(Wk, qh, wkqb);
    k_sbias  <<<2,    256, 0, stream>>>(qh, bk, sbias);
    k_flash  <<<256,  256, 0, stream>>>(matrix, mask, wkqb, sbias,
                                        scores, ctxp, mpart, spart);
    k_stats  <<<2,    256, 0, stream>>>(mpart, spart, escale, gM, gInvS);
    k_attn   <<<8192, 256, 0, stream>>>(scores, mask, gM, gInvS, attn);
    k_combine<<<2048, 256, 0, stream>>>(ctxp, escale, ctx);
    k_vo     <<<128,  256, 0, stream>>>(ctx, Wv, bv, pre);
    k_out    <<<128,  256, 0, stream>>>(pre, Wo, bo, out);
}
